// Spk_Subunit_NN_14413910246036
// MI455X (gfx1250) — matrix operand order repaired and checked by reading, not compiled, not run on hardware
//
#include <hip/hip_runtime.h>
#include <hip/hip_bf16.h>

typedef __attribute__((ext_vector_type(16))) __bf16 v16bf;
typedef __attribute__((ext_vector_type(8)))  float  v8f;
typedef __attribute__((ext_vector_type(4)))  float  f32x4;

#define T_DATA   20000
#define T_NO     200
#define SUB_NO   20
#define H_NO     10
#define COS_NO   20

#define SG_PAD    256
#define SG_STRIDE 20480   // 256 front pad + 20000 body + slack

// ---- workspace float offsets ----
#define WS_KERN   0                       // [20][400][16]  = 128000
#define WS_HK     128000                  // [20][200]      = 4000
#define WS_SPKK   132000                  // [20][200]      = 4000
#define WS_SG     136000                  // [40][20480]    = 819200
#define WS_DRIVE  955200                  // [20000][20]    = 400000
#define WS_SPK    1355200                 // [20000][20]    = 400000

// ---- d_out float offsets ----
#define O_V       0
#define O_CK      20000
#define O_HK      100000
#define O_SPK     104000
#define O_TH      504000

// 16-bit A-matrix 16x32 element -> K index (ISA 7.12.2)
__device__ __forceinline__ int a16_k(int e, int half) {
    int v = e >> 1, pos = e & 1;
    return ((v >= 4) ? 16 : 0) + 8 * half + 2 * (v & 3) + pos;
}

// ============================================================
// K1: basis, conv kernels (flipped->out, unflipped->ws), hist
//     kernel, root spike kernel, zero Sg pad.
// ============================================================
__global__ void prep_kernel(const float* __restrict__ W_conv,
                            const float* __restrict__ W_hist,
                            const float* __restrict__ Tau_root,
                            const float* __restrict__ W_root,
                            float* __restrict__ out_ck,
                            float* __restrict__ out_hk,
                            float* __restrict__ wsKern,
                            float* __restrict__ wsHk,
                            float* __restrict__ wsSpkk,
                            float* __restrict__ sg) {
    __shared__ float basis[COS_NO][T_NO];
    const float PI = 3.14159265358979323846f;
    for (int i = threadIdx.x; i < COS_NO * T_NO; i += 256) {
        int b = i / T_NO, t = i % T_NO;
        float raw = 5.f * __logf((float)t + 1.f);
        float phi = 0.5f * PI * (float)b;
        float v = 0.5f * __cosf(raw - phi) + 0.5f;
        bool in = (raw >= phi - PI) && (raw <= phi + PI);
        basis[b][t] = in ? v : 0.f;
    }
    __syncthreads();
    // conv kernels: o in [0,200), c in {0,1}, d in [0,200)
    for (int i = threadIdx.x; i < 200 * 2 * T_NO; i += 256) {
        int d = i % T_NO, c = (i / T_NO) & 1, o = i / (2 * T_NO);
        const float* wr = W_conv + (o * 2 + c) * COS_NO;
        float acc = 0.f;
        for (int b = 0; b < COS_NO; ++b) acc += wr[b] * basis[b][d];
        out_ck[(o * 2 + c) * T_NO + (T_NO - 1 - d)] = acc;      // flipped, as returned
        int s = o / H_NO, h = o % H_NO, k = c * T_NO + d;
        wsKern[(s * 400 + k) * 16 + h] = acc;                   // unflipped for conv
    }
    // zero padded rows h=10..15
    for (int i = threadIdx.x; i < SUB_NO * 400 * 6; i += 256) {
        int h = 10 + (i % 6), sk = i / 6;
        wsKern[sk * 16 + h] = 0.f;
    }
    // hist kernel = W_hist @ basis
    for (int i = threadIdx.x; i < SUB_NO * T_NO; i += 256) {
        int s = i / T_NO, d = i % T_NO;
        float acc = 0.f;
        for (int b = 0; b < COS_NO; ++b) acc += W_hist[s * COS_NO + b] * basis[b][d];
        out_hk[i] = acc;
        wsHk[i] = acc;
    }
    // root spike kernel (unflipped lag form), kern[s][0] == 0
    for (int i = threadIdx.x; i < SUB_NO * T_NO; i += 256) {
        int s = i / T_NO, d = i % T_NO;
        float tau = __expf(Tau_root[s]);
        float tt = (float)d / tau;
        wsSpkk[i] = tt * __expf(-tt) * __expf(W_root[s]);
    }
    // zero the causal left pad of the grouped signal
    for (int i = threadIdx.x; i < 40 * SG_PAD; i += 256) {
        int ch = i / SG_PAD, p = i % SG_PAD;
        sg[(size_t)ch * SG_STRIDE + p] = 0.f;
    }
}

// ============================================================
// K2: grouped synapse GEMM via WMMA bf16.
//     Sg[ch=2n+kind][t] = sum_k S[t,k] * Csyn[n,k]
//     Block = 8 waves covering 4 m-tiles x 2 n-tiles.
//     B staged zero-padded in LDS (vector fragment loads);
//     A main chunks = aligned float4 global loads (rows are
//     32B aligned: K*4 % 32 == 0 for K=1000,200);
//     A K-tail staged zero-padded in LDS. No predication in
//     the WMMA loop.
// ============================================================
__global__ void gemm_group_kernel(const float* __restrict__ S,
                                  const float* __restrict__ Csyn,
                                  float* __restrict__ sg, int K, int kind) {
    extern __shared__ __align__(32) char smem[];
    int Kpad = ((K + 31) >> 5) << 5;
    __bf16* Bl = (__bf16*)smem;              // [32][Kpad]
    __bf16* At = Bl + 32 * Kpad;             // [64][32]  (K-tail, zero padded)

    int nfull = K >> 5;
    int ktail = nfull << 5;

    // stage B
    for (int i = threadIdx.x; i < 32 * Kpad; i += 256) {
        int n = i / Kpad, k = i % Kpad;
        float v = (n < SUB_NO && k < K) ? Csyn[(size_t)n * K + k] : 0.f;
        Bl[n * Kpad + k] = (__bf16)v;
    }
    // stage A tail columns for this block's 64 time rows
    int rowbase = blockIdx.x * 64;
    for (int i = threadIdx.x; i < 64 * 32; i += 256) {
        int row = i / 32, kk = i % 32;
        int t = rowbase + row, k = ktail + kk;
        float v = (t < T_DATA && k < K) ? S[(size_t)t * K + k] : 0.f;
        At[row * 32 + kk] = (__bf16)v;
    }
    __syncthreads();

    int w = threadIdx.x >> 5;
    int lane = threadIdx.x & 31;
    int half = lane >> 4, r = lane & 15;
    int g = blockIdx.x * 8 + w;
    int mtile = g >> 1, ntile = g & 1;

    if (mtile < T_DATA / 16) {
        int t0 = mtile * 16;
        int arow = t0 + r;
        int n16 = ntile * 16 + r;
        const __bf16* brow = Bl + n16 * Kpad;
        v8f acc = {};
        for (int ck = 0; ck < nfull; ++ck) {
            int kb = ck << 5;
            // A: two aligned 32B runs per lane
            const f32x4* ap0 = (const f32x4*)(S + (size_t)arow * K + kb + 16 * half);
            const f32x4* ap1 = (const f32x4*)(S + (size_t)arow * K + kb + 16 * half + 8);
            f32x4 a0 = ap0[0], a1 = ap0[1];
            f32x4 a2 = ap1[0], a3 = ap1[1];
            v16bf a;
            #pragma unroll
            for (int e = 0; e < 4; ++e) {
                a[e]      = (__bf16)a0[e];
                a[e + 4]  = (__bf16)a1[e];
                a[e + 8]  = (__bf16)a2[e];
                a[e + 12] = (__bf16)a3[e];
            }
            v16bf b = *(const v16bf*)(brow + kb + 16 * half);
            acc = __builtin_amdgcn_wmma_f32_16x16x32_bf16(false, a, false, b,
                                                          (short)0, acc, false, false);
        }
        if (K & 31) {                        // zero-padded tail chunk from LDS
            const __bf16* atr = At + (arow - rowbase) * 32;
            v16bf a, b;
            #pragma unroll
            for (int e = 0; e < 16; ++e) a[e] = atr[16 * half + e];
            b = *(const v16bf*)(brow + ktail + 16 * half);
            acc = __builtin_amdgcn_wmma_f32_16x16x32_bf16(false, a, false, b,
                                                          (short)0, acc, false, false);
        }
        if (n16 < SUB_NO) {
            int ch = 2 * n16 + kind;
            float* dst = sg + (size_t)ch * SG_STRIDE + SG_PAD;
            #pragma unroll
            for (int j = 0; j < 8; ++j) dst[t0 + j + 8 * half] = acc[j];
        }
    }
}

// ============================================================
// K3: grouped temporal conv as Toeplitz WMMA:
//     D[t', h] = sum_{c,d} S[c, t-d] * kern[h, c, d]
//     A = signal Toeplitz (scalar LDS reads, zero-padded),
//     B = kernel transposed [h][c*224+d] bf16 (vector LDS
//     fragment loads, d in [200,224) zero). Fused epilogue:
//     leaky -> W_ff -> leaky -> W_ff2 -> drive.
//     grid = (157, SUB_NO), block = 256 (8 waves x 16 t).
// ============================================================
__global__ void conv_drive_kernel(const float* __restrict__ W_ff,
                                  const float* __restrict__ W_ff2,
                                  const float* __restrict__ threshold,
                                  const float* __restrict__ wsKern,
                                  const float* __restrict__ sg,
                                  float* __restrict__ drive) {
    __shared__ __align__(32) __bf16 kernT[16 * 448];  // [h][c*224+d], zero padded
    __shared__ float sS[2][352];                       // front pad 24 + 328 samples
    __shared__ float convD[8][16 * 16];                // [wave][t'*16 + h]
    int s = blockIdx.y;
    int tbase = blockIdx.x * 128;

    for (int i = threadIdx.x; i < 16 * 448; i += 256) {
        int h = i / 448, kx = i % 448;
        int c = kx / 224, d = kx % 224;
        float v = (d < T_NO) ? wsKern[(s * 400 + c * T_NO + d) * 16 + h] : 0.f;
        kernT[h * 448 + kx] = (__bf16)v;
    }
    for (int i = threadIdx.x; i < 2 * 352; i += 256) {
        int c = i / 352, iu = i % 352;
        int tl = iu - 24;                              // local time, -24..327
        float v = 0.f;
        if (tl >= 0 && tl < 328) {
            size_t idx = (size_t)(2 * s + c) * SG_STRIDE + SG_PAD + tbase - T_NO + tl;
            v = sg[idx];                               // max idx = 20351 < SG_STRIDE
        }
        sS[c][iu] = v;
    }
    __syncthreads();

    int w = threadIdx.x >> 5, lane = threadIdx.x & 31;
    int half = lane >> 4, r = lane & 15;
    int ubase = w * 16 + r + 224;                      // sS index = ubase - d
    v8f acc = {};
    #pragma unroll
    for (int c = 0; c < 2; ++c) {
        const float* sc = sS[c];
        #pragma unroll
        for (int ck = 0; ck < 7; ++ck) {               // 224 = 7 chunks of 32
            int db = ck * 32;
            v16bf a;
            #pragma unroll
            for (int e = 0; e < 16; ++e) {
                int d = db + 16 * half + e;
                a[e] = (__bf16)sc[ubase - d];          // iu in [1, 351]
            }
            v16bf b = *(const v16bf*)(kernT + r * 448 + c * 224 + db + 16 * half);
            acc = __builtin_amdgcn_wmma_f32_16x16x32_bf16(false, a, false, b,
                                                          (short)0, acc, false, false);
        }
    }
    #pragma unroll
    for (int j = 0; j < 8; ++j) {
        float x = acc[j];
        x = x > 0.f ? x : 0.01f * x;                   // leaky_relu(conv_out)
        int tp = j + 8 * half;                         // M = time row
        convD[w][tp * 16 + r] = x;                     // N = r = h
    }
    __syncthreads();

    if (threadIdx.x < 128) {
        int tl = (int)threadIdx.x;
        int t = tbase + tl;
        if (t < T_DATA) {
            const float* cv  = &convD[tl >> 4][(tl & 15) * 16];
            const float* wff = W_ff + s * H_NO * H_NO;
            float drv = threshold[s];
            #pragma unroll
            for (int k = 0; k < H_NO; ++k) {
                float ak = 0.f;
                #pragma unroll
                for (int h = 0; h < H_NO; ++h) ak += cv[h] * wff[h * H_NO + k];
                ak = ak > 0.f ? ak : 0.01f * ak;       // leaky_relu(conv_in)
                drv += ak * W_ff2[s * H_NO + k];
            }
            drive[t * SUB_NO + s] = drv;
        }
    }
}

// ============================================================
// K4: sequential spike scan. Subunits independent -> one wave
//     per subunit, 256-slot LDS spike ring, no barriers inside
//     the 20000-step loop.
// ============================================================
__global__ void scan_kernel(const float* __restrict__ drive,
                            const float* __restrict__ wsHk,
                            float* __restrict__ out_spk,
                            float* __restrict__ out_th,
                            float* __restrict__ wsSpk) {
    __shared__ float ring[SUB_NO][256];
    __shared__ float hkL[SUB_NO][T_NO];
    for (int i = threadIdx.x; i < SUB_NO * 256; i += 640) ring[i / 256][i % 256] = 0.f;
    for (int i = threadIdx.x; i < SUB_NO * T_NO; i += 640) hkL[i / T_NO][i % T_NO] = wsHk[i];
    __syncthreads();

    int w = threadIdx.x >> 5;      // subunit
    int lane = threadIdx.x & 31;
    for (int t = 0; t < T_DATA; ++t) {
        float acc = 0.f;
        #pragma unroll
        for (int c = 0; c < 7; ++c) {
            int tau = c * 32 + lane;
            if (tau < T_NO)
                acc += ring[w][(t + 56 + tau) & 255] * hkL[w][tau]; // (t-200+tau) mod 256
        }
        #pragma unroll
        for (int off = 16; off > 0; off >>= 1) acc += __shfl_xor(acc, off, 32);
        if (lane == 0) {
            float sub = acc + drive[t * SUB_NO + w];
            float spk = sub > 0.f ? 1.f : 0.f;
            out_th[t * SUB_NO + w]  = sub;
            out_spk[t * SUB_NO + w] = spk;
            wsSpk[t * SUB_NO + w]   = spk;
            ring[w][t & 255] = spk;
        }
        // wave-coherent: no cross-wave data -> no barrier needed
    }
}

// ============================================================
// K5: root alpha-kernel conv over spikes + sum over subunits.
//     V[t] = sum_s sum_d spikes[t-1-d, s] * spkk[s][d]
// ============================================================
__global__ void root_kernel(const float* __restrict__ wsSpk,
                            const float* __restrict__ wsSpkk,
                            float* __restrict__ outV) {
    __shared__ float sspk[456][SUB_NO];
    __shared__ float kL[SUB_NO][T_NO];
    int t0 = blockIdx.x * 256;
    for (int i = threadIdx.x; i < 456 * SUB_NO; i += 256) {
        int row = i / SUB_NO, s = i % SUB_NO;
        int t = t0 - T_NO + row;
        sspk[row][s] = (t >= 0 && t < T_DATA) ? wsSpk[t * SUB_NO + s] : 0.f;
    }
    for (int i = threadIdx.x; i < SUB_NO * T_NO; i += 256)
        kL[i / T_NO][i % T_NO] = wsSpkk[i];
    __syncthreads();
    int t = t0 + (int)threadIdx.x;
    if (t < T_DATA) {
        float V = 0.f;
        for (int s = 0; s < SUB_NO; ++s) {
            float a = 0.f;
            for (int d = 0; d < T_NO; ++d)
                a += sspk[threadIdx.x + (T_NO - 1) - d][s] * kL[s][d];
            V += a;
        }
        outV[t] = V;
    }
}

extern "C" void kernel_launch(void* const* d_in, const int* in_sizes, int n_in,
                              void* d_out, int out_size, void* d_ws, size_t ws_size,
                              hipStream_t stream) {
    const float* S_e       = (const float*)d_in[0];
    const float* S_i       = (const float*)d_in[1];
    const float* C_syn_e   = (const float*)d_in[2];
    const float* C_syn_i   = (const float*)d_in[3];
    const float* W_conv    = (const float*)d_in[4];
    const float* threshold = (const float*)d_in[5];
    const float* W_ff      = (const float*)d_in[6];
    const float* W_ff2     = (const float*)d_in[7];
    const float* Tau_root  = (const float*)d_in[8];
    const float* W_root    = (const float*)d_in[9];
    const float* W_hist    = (const float*)d_in[10];

    float* ws      = (float*)d_ws;
    float* wsKern  = ws + WS_KERN;
    float* wsHk    = ws + WS_HK;
    float* wsSpkk  = ws + WS_SPKK;
    float* wsSg    = ws + WS_SG;
    float* wsDrive = ws + WS_DRIVE;
    float* wsSpk   = ws + WS_SPK;

    float* out    = (float*)d_out;
    float* outV   = out + O_V;
    float* outCk  = out + O_CK;
    float* outHk  = out + O_HK;
    float* outSpk = out + O_SPK;
    float* outTh  = out + O_TH;

    prep_kernel<<<1, 256, 0, stream>>>(W_conv, W_hist, Tau_root, W_root,
                                       outCk, outHk, wsKern, wsHk, wsSpkk, wsSg);

    // dynamic LDS: B [32][Kpad] bf16 + A-tail [64][32] bf16
    size_t lds_e = (size_t)32 * 1024 * 2 + 64 * 32 * 2;   // 69632 B
    size_t lds_i = (size_t)32 * 224  * 2 + 64 * 32 * 2;   // 18432 B
    gemm_group_kernel<<<313, 256, lds_e, stream>>>(S_e, C_syn_e, wsSg, 1000, 0);
    gemm_group_kernel<<<313, 256, lds_i, stream>>>(S_i, C_syn_i, wsSg, 200, 1);

    conv_drive_kernel<<<dim3(157, SUB_NO), 256, 0, stream>>>(W_ff, W_ff2, threshold,
                                                             wsKern, wsSg, wsDrive);
    scan_kernel<<<1, 640, 0, stream>>>(wsDrive, wsHk, outSpk, outTh, wsSpk);
    root_kernel<<<79, 256, 0, stream>>>(wsSpk, wsSpkk, outV);
}
